// CrashingVids_73813307949074
// MI455X (gfx1250) — matrix operand order repaired and checked by reading, not compiled, not run on hardware
//
#include <hip/hip_runtime.h>
#include <math.h>
#include <stdint.h>

typedef __attribute__((ext_vector_type(16))) __bf16 v16bf;
typedef __attribute__((ext_vector_type(8)))  __bf16 v8bf;
typedef __attribute__((ext_vector_type(8)))  float  v8f;

#define T_LEN   750
#define F_DIM   2048
#define NCLU    10
#define NCLS    20
#define BATCH   16
#define EMBED   2048
#define MTOT    (BATCH * T_LEN)    // 12000
#define KTOT    (3 * F_DIM)        // 6144
#define NIT     (KTOT / 32)        // 192 (even)
#define MTILES128 ((MTOT + 127) / 128)  // 94
#define MTILES64  ((MTOT + 63) / 64)    // 188

// ---------------------------------------------------------------- utilities

__global__ void CV_init(float* normsq, __bf16* zrow) {
    int i = threadIdx.x;
    if (i < 16) normsq[i] = 0.f;
    for (int j = i; j < 2048; j += 256) zrow[j] = (__bf16)0.f;
}

__global__ void CV_cvt_bf16(const float* __restrict__ src, __bf16* __restrict__ dst, int n) {
    int i = blockIdx.x * 256 + threadIdx.x;
    if (i < n) dst[i] = (__bf16)src[i];
}

// wPackB[e][k*2048+f] = bf16(w_embed[e][f][k]);  w_embed shape (E, F, 3)
__global__ void CV_pack_wembed(const float* __restrict__ w, __bf16* __restrict__ wp) {
    long i = (long)blockIdx.x * 256 + threadIdx.x;
    if (i >= (long)EMBED * KTOT) return;
    int e   = (int)(i / KTOT);
    int rem = (int)(i % KTOT);
    int k   = rem >> 11;      // / 2048
    int f   = rem & 2047;
    wp[i] = (__bf16)w[(size_t)e * KTOT + (size_t)f * 3 + k];
}

// wClsPack[n][e], n padded 20 -> 32 with zeros; w_cls shape (20, E, 1)
__global__ void CV_pack_wcls(const float* __restrict__ w, __bf16* __restrict__ wp) {
    int i = blockIdx.x * 256 + threadIdx.x;
    if (i >= 32 * EMBED) return;
    int n = i / EMBED, e = i % EMBED;
    wp[i] = (n < NCLS) ? (__bf16)w[(size_t)n * EMBED + e] : (__bf16)0.f;
}

// ------------------------------------------------------------- distances
__global__ void CV_dist(const float* __restrict__ x, const float* __restrict__ cen,
                        float* __restrict__ d, float* __restrict__ normsq) {
    const int t   = blockIdx.x;
    const int tid = threadIdx.x;
    __shared__ float xs[16][128];
    __shared__ float cs[10][128];
    const int  b = tid / NCLU, c = tid % NCLU;
    const bool active = tid < BATCH * NCLU;
    float acc = 0.f;
    for (int f0 = 0; f0 < F_DIM; f0 += 128) {
        __syncthreads();
        for (int i = tid; i < 16 * 128; i += 256)
            xs[i >> 7][i & 127] = x[((size_t)(i >> 7) * T_LEN + t) * F_DIM + f0 + (i & 127)];
        for (int i = tid; i < 10 * 128; i += 256)
            cs[i >> 7][i & 127] = cen[((size_t)(i >> 7) * T_LEN + t) * F_DIM + f0 + (i & 127)];
        __syncthreads();
        if (active) {
            for (int j = 0; j < 128; ++j) {
                float df = xs[b][j] - cs[c][j];
                acc += df * df;
            }
        }
    }
    if (active) {
        d[((size_t)b * T_LEN + t) * NCLU + c] = acc;
        atomicAdd(&normsq[b], acc * acc);
    }
}

// ------------------------------------------------------------------ fuse
__global__ void CV_fuse(const float* __restrict__ x, const float* __restrict__ cen,
                        const float* __restrict__ scale, const float* __restrict__ d,
                        const float* __restrict__ normsq, __bf16* __restrict__ xfbf) {
    const int bt = blockIdx.x;
    const int b = bt / T_LEN, t = bt % T_LEN;
    const int tid = threadIdx.x;
    __shared__ float wc[NCLU];
    if (tid < NCLU) {
        int   widx = tid * T_LEN + t;
        int   t2 = widx / NCLU, c2 = widx % NCLU;
        float nrm = fmaxf(sqrtf(normsq[b]), 1e-12f);
        wc[tid] = d[((size_t)b * T_LEN + t2) * NCLU + c2] / nrm;
    }
    __syncthreads();
    const float s0 = scale[0];
    for (int f = tid; f < F_DIM; f += 256) {
        float info = 0.f;
        #pragma unroll
        for (int c = 0; c < NCLU; ++c)
            info += cen[((size_t)c * T_LEN + t) * F_DIM + f] * wc[c];
        float v = x[((size_t)b * T_LEN + t) * F_DIM + f] + s0 * info;
        xfbf[((size_t)b * T_LEN + t) * F_DIM + f] = (__bf16)v;
    }
}

// ------------------------------------------------- conv-as-GEMM (WMMA bf16)
// Tile 128(M) x 64(N), 256 threads (8 waves), K step 32, manual 2x unroll.
// A: GLOBAL_LOAD_ASYNC_TO_LDS_B128, triple-buffered LDS, issue distance 2,
//    rotating byte-offset registers (no % in the hot loop).
// B: explicit ping-pong register prefetch (b0/b1), no inter-stage copies.
// Async issue sits between the fragment ds_loads and the WMMAs to cover
// LDS latency. Padded conv rows point at a zero workspace row.
__launch_bounds__(256)
__global__ void CV_gemm_embed(const __bf16* __restrict__ A, const __bf16* __restrict__ B,
                              const float* __restrict__ bias, const __bf16* __restrict__ zrow,
                              __bf16* __restrict__ H) {
    __shared__ alignas(128) __bf16 As[3][128][32];
    const int tid  = threadIdx.x;
    const int lane = tid & 31, wv = tid >> 5;
    const int m0 = blockIdx.x * 128;
    const int e0 = blockIdx.y * 64;
    const int l  = lane & 15, hi = lane >> 4;
    const int q  = tid & 3;                 // 16B chunk within a 64B row
    const int rr[2] = { tid >> 2, (tid >> 2) + 64 };

    // per (chunk, conv-phase k) source pointer; padded rows -> zero row
    const __bf16* srcBase[2][3];
    #pragma unroll
    for (int i = 0; i < 2; ++i) {
        int m = m0 + rr[i];
        bool mv = (m < MTOT);
        int b = 0, t = 0;
        if (mv) { b = m / T_LEN; t = m % T_LEN; }
        const __bf16* rowStart = A + ((long)b * T_LEN + t) * F_DIM;
        #pragma unroll
        for (int k = 0; k < 3; ++k) {
            int  tt = t + k - 1;
            bool ok = mv && ((unsigned)tt < (unsigned)T_LEN);
            srcBase[i][k] = ok ? (rowStart + (long)(k - 1) * F_DIM) : zrow;
        }
    }

    const unsigned asBase  = (unsigned)(uintptr_t)&As[0][0][0];  // LDS byte offset
    const unsigned ldsThr0 = asBase + (unsigned)(rr[0] * 64 + q * 16);
    const unsigned ldsThr1 = asBase + (unsigned)(rr[1] * 64 + q * 16);

    auto issueA = [&](int itx, unsigned bufOff) {
        const int kk0 = itx * 32;
        const int k   = kk0 >> 11;
        const int f0  = kk0 & 2047;
        const __bf16* g0 = srcBase[0][0];
        const __bf16* g1 = srcBase[1][0];
        if (k == 1) { g0 = srcBase[0][1]; g1 = srcBase[1][1]; }
        if (k == 2) { g0 = srcBase[0][2]; g1 = srcBase[1][2]; }
        g0 += f0 + q * 8;
        g1 += f0 + q * 8;
        const unsigned l0 = ldsThr0 + bufOff;
        const unsigned l1 = ldsThr1 + bufOff;
        asm volatile("global_load_async_to_lds_b128 %0, %1, off"
                     :: "v"(l0), "v"(g0) : "memory");
        asm volatile("global_load_async_to_lds_b128 %0, %1, off"
                     :: "v"(l1), "v"(g1) : "memory");
    };

    v8f acc[4] = {};
    const __bf16* bp = B + (size_t)(e0 + l) * KTOT + (hi << 4);

    auto loadB = [&](v16bf (&dst)[4], int itx) {
        const int kk = itx * 32;
        #pragma unroll
        for (int ct = 0; ct < 4; ++ct)
            dst[ct] = *(const v16bf*)(bp + (size_t)ct * 16 * KTOT + kk);
    };

    auto stage = [&](const v16bf (&bc)[4], unsigned curO,
                     bool doIssue, int issIt, unsigned issO, bool lastWait) {
        if (!lastWait) asm volatile("s_wait_asynccnt 0x2" ::: "memory");
        else           asm volatile("s_wait_asynccnt 0x0" ::: "memory");
        __syncthreads();                     // everyone's tile data visible
        union { v16bf v; v8bf h[2]; } af;
        const char* fb = (const char*)&As[0][0][0] + curO + (16 * wv + l) * 64;
        af.h[0] = *(const v8bf*)(fb + hi * 32);
        af.h[1] = *(const v8bf*)(fb + hi * 32 + 16);
        if (doIssue) issueA(issIt, issO);    // covers LDS latency before WMMAs
        #pragma unroll
        for (int ct = 0; ct < 4; ++ct)
            acc[ct] = __builtin_amdgcn_wmma_f32_16x16x32_bf16(
                false, af.v, false, bc[ct], (short)0, acc[ct], false, false);
    };

    v16bf b0[4], b1[4];
    loadB(b0, 0);
    issueA(0, 0u);
    issueA(1, 8192u);
    unsigned curO = 0u, issO = 16384u;       // rotating triple-buffer offsets

    for (int it = 0; it < NIT; it += 2) {
        loadB(b1, it + 1);                                   // it+1 <= 191 always
        stage(b0, curO, (it + 2 < NIT), it + 2, issO, false);
        curO = (curO == 16384u) ? 0u : curO + 8192u;
        issO = (issO == 16384u) ? 0u : issO + 8192u;

        if (it + 2 < NIT) loadB(b0, it + 2);
        stage(b1, curO, (it + 3 < NIT), it + 3, issO, (it + 1 == NIT - 1));
        curO = (curO == 16384u) ? 0u : curO + 8192u;
        issO = (issO == 16384u) ? 0u : issO + 8192u;
    }

    #pragma unroll
    for (int ct = 0; ct < 4; ++ct) {
        const int n   = e0 + ct * 16 + l;
        const float bv = bias[n];
        #pragma unroll
        for (int rI = 0; rI < 8; ++rI) {
            int m = m0 + 16 * wv + rI + hi * 8;
            if (m < MTOT) {
                float v = acc[ct][rI] + bv;
                H[(size_t)m * EMBED + n] = (__bf16)(v > 0.f ? v : 0.f);
            }
        }
    }
}

// --------------------------------------------- class conv (WMMA bf16, N=32)
__launch_bounds__(128)
__global__ void CV_gemm_cls(const __bf16* __restrict__ A, const __bf16* __restrict__ B,
                            float* __restrict__ casOut) {
    __shared__ alignas(128) __bf16 As[64][32];
    const int tid  = threadIdx.x;
    const int lane = tid & 31, wv = tid >> 5;
    const int m0 = blockIdx.x * 64;
    const int l  = lane & 15, hi = lane >> 4;

    const int q = tid & 3;
    int  rr[2] = { tid >> 2, (tid >> 2) + 32 };
    long rowBase[2]; bool rowValid[2];
    #pragma unroll
    for (int i = 0; i < 2; ++i) {
        int m = m0 + rr[i];
        rowValid[i] = (m < MTOT);
        rowBase[i]  = rowValid[i] ? (long)m * EMBED : 0;
    }

    v8f acc[2] = {};
    const __bf16* bp = B + (size_t)l * EMBED + (hi << 4);

    for (int kk0 = 0; kk0 < EMBED; kk0 += 32) {
        uint4 av[2];
        #pragma unroll
        for (int i = 0; i < 2; ++i) {
            if (rowValid[i]) av[i] = *((const uint4*)(A + rowBase[i] + kk0) + q);
            else             av[i] = make_uint4(0u, 0u, 0u, 0u);
        }
        __syncthreads();
        *((uint4*)&As[rr[0]][0] + q) = av[0];
        *((uint4*)&As[rr[1]][0] + q) = av[1];
        __syncthreads();

        union { v16bf v; v8bf h[2]; } af;
        af.h[0] = *(const v8bf*)&As[16 * wv + l][hi * 16];
        af.h[1] = *(const v8bf*)&As[16 * wv + l][hi * 16 + 8];
        v16bf bf0 = *(const v16bf*)(bp + kk0);
        v16bf bf1 = *(const v16bf*)(bp + (size_t)16 * EMBED + kk0);
        acc[0] = __builtin_amdgcn_wmma_f32_16x16x32_bf16(
            false, af.v, false, bf0, (short)0, acc[0], false, false);
        acc[1] = __builtin_amdgcn_wmma_f32_16x16x32_bf16(
            false, af.v, false, bf1, (short)0, acc[1], false, false);
    }

    #pragma unroll
    for (int ct = 0; ct < 2; ++ct) {
        const int n = ct * 16 + l;
        if (n >= NCLS) continue;
        #pragma unroll
        for (int rI = 0; rI < 8; ++rI) {
            int m = m0 + 16 * wv + rI + hi * 8;
            if (m < MTOT) {
                float v = acc[ct][rI];
                casOut[(size_t)m * NCLS + n] = v > 0.f ? v : 0.f;
            }
        }
    }
}

// --------------------------------------------------------------- epilogues
__global__ void CV_act(const float* __restrict__ cas, float* __restrict__ act) {
    int m = blockIdx.x * 256 + threadIdx.x;
    if (m < MTOT) {
        float s = 0.f;
        #pragma unroll
        for (int n = 0; n < NCLS; ++n) s += cas[(size_t)m * NCLS + n];
        act[m] = s;
    }
}

__global__ void CV_vidscore(const float* __restrict__ cas, float* __restrict__ vs) {
    __shared__ float sv[NCLS];
    const int b = blockIdx.x, n = threadIdx.x;
    if (n < NCLS) {
        float s = 0.f;
        for (int t = 0; t < T_LEN; ++t) s += cas[((size_t)b * T_LEN + t) * NCLS + n];
        sv[n] = s / (float)T_LEN;
    }
    __syncthreads();
    if (n < NCLS) {
        float mx = -1e30f;
        for (int j = 0; j < NCLS; ++j) mx = fmaxf(mx, sv[j]);
        float den = 0.f;
        for (int j = 0; j < NCLS; ++j) den += __expf(sv[j] - mx);
        vs[(size_t)b * NCLS + n] = __expf(sv[n] - mx) / den;
    }
}

// ------------------------------------------------------------------- host

extern "C" void kernel_launch(void* const* d_in, const int* in_sizes, int n_in,
                              void* d_out, int out_size, void* d_ws, size_t ws_size,
                              hipStream_t stream) {
    (void)in_sizes; (void)n_in; (void)out_size; (void)ws_size;
    const float* x       = (const float*)d_in[0];
    const float* centers = (const float*)d_in[1];
    const float* scale   = (const float*)d_in[2];
    const float* w_embed = (const float*)d_in[3];
    const float* b_embed = (const float*)d_in[4];
    const float* w_cls   = (const float*)d_in[5];
    float* out = (float*)d_out;

    char*  ws  = (char*)d_ws;
    size_t off = 0;
    auto carve = [&](size_t bytes) { size_t o = off; off = (off + bytes + 255) & ~(size_t)255; return o; };
    __bf16* xbf      = (__bf16*)(ws + carve((size_t)MTOT * F_DIM * 2));
    __bf16* xfbf     = (__bf16*)(ws + carve((size_t)MTOT * F_DIM * 2));
    __bf16* wPackB   = (__bf16*)(ws + carve((size_t)EMBED * KTOT * 2));
    __bf16* wClsPack = (__bf16*)(ws + carve((size_t)32 * EMBED * 2));
    float*  dbuf     = (float*) (ws + carve((size_t)BATCH * T_LEN * NCLU * 4));
    float*  normsq   = (float*) (ws + carve(16 * 4));
    __bf16* zrow     = (__bf16*)(ws + carve(2048 * 2));
    __bf16* H        = (__bf16*)(ws + carve((size_t)MTOT * EMBED * 2));

    float* vid_f  = out;                       // (16,20)
    float* act_f  = out + 320;                 // (16,750)
    float* cas_f  = out + 12320;               // (16,750,20)
    float* vid_b  = out + 252320;              // (16,20)
    float* act_b  = out + 252640;              // (16,750)
    float* cas_b  = out + 264640;              // (16,750,20)

    const int nX = MTOT * F_DIM;

    CV_init<<<1, 256, 0, stream>>>(normsq, zrow);
    CV_cvt_bf16<<<(nX + 255) / 256, 256, 0, stream>>>(x, xbf, nX);
    CV_pack_wembed<<<(int)(((long)EMBED * KTOT + 255) / 256), 256, 0, stream>>>(w_embed, wPackB);
    CV_pack_wcls<<<(32 * EMBED + 255) / 256, 256, 0, stream>>>(w_cls, wClsPack);

    CV_dist<<<T_LEN, 256, 0, stream>>>(x, centers, dbuf, normsq);
    CV_fuse<<<MTOT, 256, 0, stream>>>(x, centers, scale, dbuf, normsq, xfbf);

    // base path (uses x)
    CV_gemm_embed<<<dim3(MTILES128, EMBED / 64), 256, 0, stream>>>(xbf, wPackB, b_embed, zrow, H);
    CV_gemm_cls<<<dim3(MTILES64, 1), 128, 0, stream>>>(H, wClsPack, cas_b);
    CV_act<<<(MTOT + 255) / 256, 256, 0, stream>>>(cas_b, act_b);
    CV_vidscore<<<BATCH, 32, 0, stream>>>(cas_b, vid_b);

    // fused path (uses x_fused)
    CV_gemm_embed<<<dim3(MTILES128, EMBED / 64), 256, 0, stream>>>(xfbf, wPackB, b_embed, zrow, H);
    CV_gemm_cls<<<dim3(MTILES64, 1), 128, 0, stream>>>(H, wClsPack, cas_f);
    CV_act<<<(MTOT + 255) / 256, 256, 0, stream>>>(cas_f, act_f);
    CV_vidscore<<<BATCH, 32, 0, stream>>>(cas_f, vid_f);
}
